// STPatchEmbedding_81990925681100
// MI455X (gfx1250) — matrix operand order repaired and checked by reading, compiled, not run on hardware
//
#include <hip/hip_runtime.h>
#include <stdint.h>

#define N_NODES 1024
#define K_NEIGH 5
#define L_TIME  2016
#define PATCH   12
#define NPATCH  168      // L_TIME / PATCH
#define EMBED   96
#define CIN     6        // (K_NEIGH+1) * IN_CH
#define KDIM    72       // CIN * PATCH
#define KPITCH  98       // LDS k-pitch (>= 96 padded K, odd dword count for banks)
#define PPAD    176      // 11 tiles of 16 columns
#define THREADS 256
#define NWAVES  8
#define CHUNKS  (CIN * L_TIME / 4)   // 3024 x 16B async chunks

typedef __attribute__((ext_vector_type(16))) _Float16 v16h;
typedef __attribute__((ext_vector_type(8)))  float    v8f;

__device__ __forceinline__ unsigned hash32(unsigned x) {
  x ^= x >> 17; x *= 0xed5ad4bbu;
  x ^= x >> 11; x *= 0xac4c1b51u;
  x ^= x >> 15; x *= 0x31848babu;
  x ^= x >> 14;
  return x;
}

// Async global->LDS 16B copy (CDNA5, tracked on ASYNCcnt, no VGPR round-trip).
__device__ __forceinline__ void async_copy_b128(unsigned lds_off, const void* gaddr) {
  asm volatile("global_load_async_to_lds_b128 %0, %1, off"
               :: "v"(lds_off), "v"(gaddr) : "memory");
}
__device__ __forceinline__ void wait_asynccnt0() {
  asm volatile("s_wait_asynccnt 0" ::: "memory");
}

// Kernel 1: Gumbel-max multinomial neighbor sampling; one block per node.
__global__ void __launch_bounds__(THREADS)
sample_neighbors_kernel(const float* __restrict__ adj, int* __restrict__ smp) {
  __shared__ float sv[THREADS];
  __shared__ int   si[THREADS];
  const int n = blockIdx.x;
  const int tid = threadIdx.x;
  for (int k = 0; k < K_NEIGH; ++k) {
    float best = -1e30f; int bidx = 0;
    for (int j = tid; j < N_NODES; j += THREADS) {
      unsigned h = hash32((unsigned)((n * K_NEIGH + k) * N_NODES + j) ^ 0x9E3779B9u);
      float u = (float)(h >> 9) * (1.0f / 8388608.0f) + 1e-7f;   // (0,1)
      float g = -logf(-logf(u));                                  // Gumbel(0,1)
      float s = logf(adj[n * N_NODES + j]) + g;
      if (s > best) { best = s; bidx = j; }
    }
    sv[tid] = best; si[tid] = bidx;
    __syncthreads();
    for (int s = THREADS / 2; s > 0; s >>= 1) {
      if (tid < s && sv[tid + s] > sv[tid]) { sv[tid] = sv[tid + s]; si[tid] = si[tid + s]; }
      __syncthreads();
    }
    if (tid == 0) smp[n * K_NEIGH + k] = si[0];
    __syncthreads();
  }
}

// Kernel 2: per (b,n) async gather + patchify conv as WMMA GEMM
//           D[96,168] = W[96,72] @ X[72,168]  (+bias)
__global__ void __launch_bounds__(THREADS)
patch_embed_kernel(const float* __restrict__ hist,
                   const int*   __restrict__ smp,
                   const float* __restrict__ conv_w,
                   const float* __restrict__ conv_b,
                   float*       __restrict__ out) {
  __shared__ alignas(16) float    Sg[CIN * L_TIME];    // raw f32 gather staging (async DMA dest)
  __shared__ alignas(16) _Float16 Wl[EMBED * KPITCH];  // [o][k], k zero-padded to 96
  __shared__ alignas(16) _Float16 Xl[PPAD  * KPITCH];  // [p][k], p zero-padded to 176
  __shared__ float Bl[EMBED];

  const int tid = threadIdx.x;
  const int b = blockIdx.x >> 10;
  const int n = blockIdx.x & (N_NODES - 1);

  // 1) Kick off async DMA of own series + 5 sampled neighbors into LDS staging.
  for (int i = tid; i < CHUNKS; i += THREADS) {
    const int c = i / (L_TIME / 4);
    const int q = i - c * (L_TIME / 4);            // 16B chunk within channel
    const int src = (c == 0) ? n : smp[n * K_NEIGH + (c - 1)];
    const float* g = hist + (size_t)(b * N_NODES + src) * L_TIME + q * 4;
    const unsigned lds_off = (unsigned)(size_t)&Sg[c * L_TIME + q * 4];
    async_copy_b128(lds_off, g);
  }

  // 2) Overlap with DMA: zero-fill padded f16 tiles, stage weights + bias.
  {
    uint32_t* wz = (uint32_t*)Wl;
    for (int i = tid; i < EMBED * KPITCH / 2; i += THREADS) wz[i] = 0u;
    uint32_t* xz = (uint32_t*)Xl;
    for (int i = tid; i < PPAD * KPITCH / 2; i += THREADS) xz[i] = 0u;
  }
  if (tid < EMBED) Bl[tid] = conv_b[tid];
  for (int i = tid; i < EMBED * KDIM; i += THREADS) {
    int o = i / KDIM, k = i - o * KDIM;
    Wl[o * KPITCH + k] = (_Float16)conv_w[i];       // flat k = c*PATCH + t matches GEMM K
  }

  // 3) Drain DMA, make staged data visible to all waves.
  wait_asynccnt0();
  __syncthreads();

  // 4) LDS->LDS transpose + f32->f16 convert into [p][k] layout.
  for (int i = tid; i < CIN * L_TIME; i += THREADS) {
    const int c = i / L_TIME;
    const int r = i - c * L_TIME;
    const int p = r / PATCH, t = r - p * PATCH;
    Xl[p * KPITCH + c * PATCH + t] = (_Float16)Sg[i];
  }
  __syncthreads();

  const int wave = tid >> 5;
  const int lane = tid & 31;
  const int li   = lane & 15;
  const int hi   = lane >> 4;

  // 6 M-tiles x 11 N-tiles = 66 tiles over 8 waves.
  for (int tile = wave; tile < 66; tile += NWAVES) {
    const int mt = tile / 11;
    const int nt = tile - mt * 11;
    const _Float16* wrow = &Wl[(mt * 16 + li) * KPITCH];
    const _Float16* xrow = &Xl[(nt * 16 + li) * KPITCH];
    v8f acc = {};
#pragma unroll
    for (int kc = 0; kc < 3; ++kc) {
      const int kb = kc * 32;
      v16h a, bm;
#pragma unroll
      for (int j = 0; j < 8; ++j) {
        // A 16x32 f16 layout: VGPR j holds K = (j/4)*16 + (j%4)*2 (+8 for lanes 16-31)
        const int ka = kb + ((j >> 2) << 4) + ((j & 3) << 1) + (hi << 3);
        a[2 * j]     = wrow[ka];
        a[2 * j + 1] = wrow[ka + 1];
        // B 32x16 f16 layout: VGPR j holds K = (j/4)*16 + (j%4)*2 (+8 for lanes 16-31)
        const int kq = kb + ((j >> 2) << 4) + ((j & 3) << 1) + (hi << 3);
        bm[2 * j]     = xrow[kq];
        bm[2 * j + 1] = xrow[kq + 1];
      }
      acc = __builtin_amdgcn_wmma_f32_16x16x32_f16(false, a, false, bm,
                                                   (short)0, acc, false, false);
    }
    // D 16x16 f32 layout: VGPR r -> M = r + 8*hi, lane%16 -> N.
    const int p = nt * 16 + li;
    if (p < NPATCH) {
      const int o0 = mt * 16 + hi * 8;
      float* op = out + ((size_t)(b * N_NODES + n) * EMBED + o0) * NPATCH + p;
#pragma unroll
      for (int r = 0; r < 8; ++r)
        op[(size_t)r * NPATCH] = acc[r] + Bl[o0 + r];
    }
  }
}

extern "C" void kernel_launch(void* const* d_in, const int* in_sizes, int n_in,
                              void* d_out, int out_size, void* d_ws, size_t ws_size,
                              hipStream_t stream) {
  const float* hist = (const float*)d_in[0];   // [B, N, 1, L] f32
  const float* adj  = (const float*)d_in[1];   // [N, N] f32
  const float* w    = (const float*)d_in[2];   // [96, 6, 12] f32
  const float* bia  = (const float*)d_in[3];   // [96] f32
  float* out = (float*)d_out;                  // [B, N, 96, 168] f32
  int* smp = (int*)d_ws;                       // [N, K_NEIGH] int scratch

  const int B = in_sizes[0] / (N_NODES * L_TIME);

  sample_neighbors_kernel<<<N_NODES, THREADS, 0, stream>>>(adj, smp);
  patch_embed_kernel<<<B * N_NODES, THREADS, 0, stream>>>(hist, smp, w, bia, out);
}
